// HQQSVDLinear_62268435857804
// MI455X (gfx1250) — matrix operand order repaired and checked by reading, not compiled, not run on hardware
//
#include <hip/hip_runtime.h>
#include <hip/hip_bf16.h>

// ---------------------------------------------------------------------------
// HQQ int4 dequant + rank-32 SVD correction + GEMM (x @ W_f^T + bias)
//   M = B*S = 8192, N = OUT = 4096, K = IN = 4096, RANK = 32, NG=32, GS=128
// Plan:
//   k1: X fp32 -> bf16 (workspace)
//   k2: W_f[N,K] = (W_q - zp)*scale + svd_up@svd_down  -> bf16 (workspace)
//   k3: WMMA bf16 GEMM, async global->LDS, 2-deep pipeline / 3 LDS buffers
// ---------------------------------------------------------------------------

typedef unsigned short u16;
typedef __attribute__((ext_vector_type(16))) __bf16 v16bf;
typedef __attribute__((ext_vector_type(8)))  __bf16 v8bf;
typedef __attribute__((ext_vector_type(8)))  float  v8f;
typedef __attribute__((ext_vector_type(8)))  unsigned short u16x8;

#define M_DIM 8192
#define N_DIM 4096
#define K_DIM 4096
#define RANK  32
#define NGRP  32
#define GSIZE 128

static __device__ __forceinline__ u16 f2bf(float f) {
    union { float f; unsigned u; } v; v.f = f;
    unsigned u = v.u;
    unsigned r = u + 0x7FFFu + ((u >> 16) & 1u);   // RNE truncate to bf16
    return (u16)(r >> 16);
}

// Async 16B global->LDS copy (CDNA5).  GVS mode: mem = SGPR_base + voff.
static __device__ __forceinline__ void async_ld16(unsigned lds_off,
                                                  unsigned voff,
                                                  const void* sbase) {
    asm volatile("global_load_async_to_lds_b128 %0, %1, %2"
                 :
                 : "v"(lds_off), "v"(voff), "s"(sbase)
                 : "memory");
}
// Async loads complete in order; each thread issues 4 per tile, so
// ASYNCcnt<=4 means the oldest in-flight tile has fully landed in LDS.
static __device__ __forceinline__ void wait_async0() {
    asm volatile("s_wait_asynccnt 0x0" ::: "memory");
}
static __device__ __forceinline__ void wait_async4() {
    asm volatile("s_wait_asynccnt 0x4" ::: "memory");
}

// ---------------------------------------------------------------------------
// Kernel 1: fp32 -> bf16 conversion of X (8 elems / thread)
// ---------------------------------------------------------------------------
__global__ __launch_bounds__(256) void cvt_x_bf16(const float* __restrict__ x,
                                                  u16* __restrict__ xh) {
    size_t idx = ((size_t)blockIdx.x * 256 + threadIdx.x) * 8;
    float4 a = *(const float4*)(x + idx);
    float4 b = *(const float4*)(x + idx + 4);
    u16x8 v;
    v[0] = f2bf(a.x); v[1] = f2bf(a.y); v[2] = f2bf(a.z); v[3] = f2bf(a.w);
    v[4] = f2bf(b.x); v[5] = f2bf(b.y); v[6] = f2bf(b.z); v[7] = f2bf(b.w);
    *(u16x8*)(xh + idx) = v;
}

// ---------------------------------------------------------------------------
// Kernel 2: W_f = dequant(W_q) + svd_up @ svd_down  -> bf16 [N_DIM x K_DIM]
// Block tile: 32 output rows x 128 cols (one quant group per row).
// ---------------------------------------------------------------------------
__global__ __launch_bounds__(256) void dequant_svd_bf16(
        const int*   __restrict__ Wq,
        const float* __restrict__ up,
        const float* __restrict__ down,
        const float* __restrict__ scale,
        const float* __restrict__ zp,
        u16*         __restrict__ Wh) {
    __shared__ float downs[RANK][GSIZE];
    __shared__ float ups[32][RANK];

    const int t  = threadIdx.x;
    const int i0 = blockIdx.x * GSIZE;
    const int o0 = blockIdx.y * 32;

#pragma unroll
    for (int it = 0; it < 4; ++it) {
        int ci = it * 256 + t;
        int r = ci >> 5, c = (ci & 31) * 4;
        *(float4*)&downs[r][c] = *(const float4*)(down + (size_t)r * K_DIM + i0 + c);
    }
    {
        int r = t >> 3, c = (t & 7) * 4;
        *(float4*)&ups[r][c] = *(const float4*)(up + (size_t)(o0 + r) * RANK + c);
    }
    __syncthreads();

    const int ro = t >> 3;
    const int cb = (t & 7) << 4;

    float acc[16];
#pragma unroll
    for (int j = 0; j < 16; ++j) acc[j] = 0.0f;
#pragma unroll 4
    for (int r = 0; r < RANK; ++r) {
        float u = ups[ro][r];
#pragma unroll
        for (int j = 0; j < 16; ++j) acc[j] += u * downs[r][cb + j];
    }

    const int o = o0 + ro;
    const int g = blockIdx.x;
    const float s = scale[o * NGRP + g];
    const float z = zp[o * NGRP + g];
    const int* wq = Wq + ((size_t)o * NGRP + g) * GSIZE + cb;

    u16x8 p0, p1;
#pragma unroll
    for (int j = 0; j < 8; ++j)
        p0[j] = f2bf(((float)wq[j] - z) * s + acc[j]);
#pragma unroll
    for (int j = 0; j < 8; ++j)
        p1[j] = f2bf(((float)wq[8 + j] - z) * s + acc[8 + j]);

    u16* dst = Wh + (size_t)o * K_DIM + i0 + cb;
    *(u16x8*)dst = p0;
    *(u16x8*)(dst + 8) = p1;
}

// ---------------------------------------------------------------------------
// Kernel 3: C[M,N] = Xh[M,K] * Wh[N,K]^T + bias, bf16 WMMA 16x16x32
// Block 128x128, 256 threads = 8 waves (2x4), 64x32 per wave, BK=32.
// 2-deep async pipeline over 3 LDS buffers (61.4 KB total).
// ---------------------------------------------------------------------------
#define BM 128
#define BN 128
#define BK 32
#define BKP 40   // +8 halves pad: row stride 80B (16B aligned, bank spread)

__global__ __launch_bounds__(256) void gemm_bf16_wmma(
        const u16*  __restrict__ Xh,
        const u16*  __restrict__ Wh,
        const float* __restrict__ bias,
        float*       __restrict__ out) {
    __shared__ __attribute__((aligned(16))) u16 As[3][BM][BKP];
    __shared__ __attribute__((aligned(16))) u16 Bs[3][BN][BKP];

    const int tid  = threadIdx.x;
    const int lane = tid & 31;
    const int wid  = tid >> 5;
    const int wm   = wid >> 2;           // 0..1 -> 64-row slab
    const int wn   = wid & 3;            // 0..3 -> 32-col slab
    const int l16  = lane & 15;
    const int lhi  = lane >> 4;          // 0 or 1
    const int m0   = blockIdx.y * BM;
    const int n0   = blockIdx.x * BN;

    v8f acc[4][2];
#pragma unroll
    for (int mi = 0; mi < 4; ++mi)
#pragma unroll
        for (int ni = 0; ni < 2; ++ni)
#pragma unroll
            for (int r = 0; r < 8; ++r) acc[mi][ni][r] = 0.0f;

    // This thread's two 16B chunks of each 128x32 tile (chunk = 8 halves).
    // Global byte offsets are loop-invariant; tile base advances in SGPRs.
    const int r0 = tid >> 2,            c0 = (tid & 3) * 8;
    const int r1 = (256 + tid) >> 2,    c1 = (tid & 3) * 8;
    const unsigned gv0 = (unsigned)(r0 * K_DIM + c0) * 2u;
    const unsigned gv1 = (unsigned)(r1 * K_DIM + c1) * 2u;

    auto issueTile = [&](int kt, int buf) {
        const u16* aSrc = Xh + (size_t)m0 * K_DIM + kt * BK;
        const u16* bSrc = Wh + (size_t)n0 * K_DIM + kt * BK;
        async_ld16((unsigned)(uintptr_t)&As[buf][r0][c0], gv0, aSrc);
        async_ld16((unsigned)(uintptr_t)&As[buf][r1][c1], gv1, aSrc);
        async_ld16((unsigned)(uintptr_t)&Bs[buf][r0][c0], gv0, bSrc);
        async_ld16((unsigned)(uintptr_t)&Bs[buf][r1][c1], gv1, bSrc);
    };

    const int KT = K_DIM / BK;           // 128
    issueTile(0, 0);
    issueTile(1, 1);

    int cur = 0;                         // kt % 3
    for (int kt = 0; kt < KT; ++kt) {
        // Tiles kt (and kt+1 unless at the tail) are in flight; ensure the
        // oldest (kt) has landed, then make it visible to all waves.
        if (kt + 1 < KT) wait_async4(); else wait_async0();
        __syncthreads();

        if (kt + 2 < KT) {
            const int ib = (cur + 2 >= 3) ? cur - 1 : cur + 2;   // (kt+2)%3
            issueTile(kt + 2, ib);
        }
        if (kt + 4 < KT) {
            // L2 prefetch four tiles ahead (global_prefetch_b8)
            __builtin_prefetch(Xh + (size_t)(m0 + r0) * K_DIM + (kt + 4) * BK + c0, 0, 1);
            __builtin_prefetch(Wh + (size_t)(n0 + r0) * K_DIM + (kt + 4) * BK + c0, 0, 1);
        }

        // B fragments: lane = column n (l16), K[0..7]+K[16..23], +8 for lane>=16
        v16bf bfrag[2];
#pragma unroll
        for (int ni = 0; ni < 2; ++ni) {
            const u16* p = &Bs[cur][wn * 32 + ni * 16 + l16][lhi * 8];
            v8bf b0 = *(const v8bf*)p;
            v8bf b1 = *(const v8bf*)(p + 16);
            bfrag[ni] = __builtin_shufflevector(b0, b1,
                0, 1, 2, 3, 4, 5, 6, 7, 8, 9, 10, 11, 12, 13, 14, 15);
        }

#pragma unroll
        for (int mi = 0; mi < 4; ++mi) {
            // A fragment: lane<16 -> K[0..7]+K[16..23]; lane>=16 -> +8
            const u16* p = &As[cur][wm * 64 + mi * 16 + l16][lhi * 8];
            v8bf a0 = *(const v8bf*)p;
            v8bf a1 = *(const v8bf*)(p + 16);
            v16bf afrag = __builtin_shufflevector(a0, a1,
                0, 1, 2, 3, 4, 5, 6, 7, 8, 9, 10, 11, 12, 13, 14, 15);
#pragma unroll
            for (int ni = 0; ni < 2; ++ni) {
                acc[mi][ni] = __builtin_amdgcn_wmma_f32_16x16x32_bf16(
                    false, afrag, false, bfrag[ni],
                    (short)0, acc[mi][ni], false, false);
            }
        }

        cur = (cur == 2) ? 0 : cur + 1;
    }

    // Epilogue: C layout -> lanes 0-15 rows r, lanes 16-31 rows r+8
#pragma unroll
    for (int ni = 0; ni < 2; ++ni) {
        const int col = n0 + wn * 32 + ni * 16 + l16;
        const float bv = bias[col];
#pragma unroll
        for (int mi = 0; mi < 4; ++mi) {
            const int rowBase = m0 + wm * 64 + mi * 16 + lhi * 8;
#pragma unroll
            for (int r = 0; r < 8; ++r)
                out[(size_t)(rowBase + r) * N_DIM + col] = acc[mi][ni][r] + bv;
        }
    }
}

// ---------------------------------------------------------------------------
// Host launcher
// ---------------------------------------------------------------------------
extern "C" void kernel_launch(void* const* d_in, const int* in_sizes, int n_in,
                              void* d_out, int out_size, void* d_ws, size_t ws_size,
                              hipStream_t stream) {
    const float* x        = (const float*)d_in[0];
    const int*   W_q      = (const int*)  d_in[1];
    const float* svd_up   = (const float*)d_in[2];
    const float* svd_down = (const float*)d_in[3];
    const float* scale    = (const float*)d_in[4];
    const float* zp       = (const float*)d_in[5];
    const float* bias     = (const float*)d_in[6];
    float* out = (float*)d_out;

    u16* Xh = (u16*)d_ws;                          // [M,K] bf16
    u16* Wh = Xh + (size_t)M_DIM * K_DIM;          // [N,K] bf16

    cvt_x_bf16<<<((size_t)M_DIM * K_DIM) / (256 * 8), 256, 0, stream>>>(x, Xh);

    dequant_svd_bf16<<<dim3(K_DIM / GSIZE, N_DIM / 32), 256, 0, stream>>>(
        W_q, svd_up, svd_down, scale, zp, Wh);

    gemm_bf16_wmma<<<dim3(N_DIM / BN, M_DIM / BM), 256, 0, stream>>>(
        Xh, Wh, bias, out);
}
